// DhmmLm_39659728011982
// MI455X (gfx1250) — matrix operand order repaired and checked by reading, not compiled, not run on hardware
//
#include <hip/hip_runtime.h>
#include <cstddef>

// Problem constants (match reference)
#define Vv 10000
#define Cc 1024
#define Ss 128
#define Hh 256
#define Bb 16
#define Tt 128
#define VW ((Vv + 31) / 32)   // 313 words of mask bits per state row
#define KT (Hh / 32)          // 8 K-tiles of 32 for every GEMM (K = H = 256)

typedef __attribute__((ext_vector_type(16))) __bf16 v16bf;
typedef __attribute__((ext_vector_type(8)))  float  v8f;

__device__ __forceinline__ __bf16 bf_from_f32(float f) {
  unsigned u = __builtin_bit_cast(unsigned, f);
  u += 0x7FFFu + ((u >> 16) & 1u);                       // round-to-nearest-even
  return __builtin_bit_cast(__bf16, (unsigned short)(u >> 16));
}

// ---- packed WMMA operand layouts -------------------------------------------------
// A (MxK, 16x32 tiles): fragment element e of lane l maps to (m = l&15,
//   klocal = f(l>>4, e)) with f per ISA "16-bit A-Matrix 16x32" table.
// Packed index: ((mt*(K/32)+kt)*32 + lane)*16 + e  -> each lane's 16 bf16 contiguous.
__device__ __forceinline__ size_t apack_idx(int m, int k, int K) {
  int kt = k >> 5, kl = k & 31;
  int h  = (kl >> 3) & 1;
  int p  = (kl < 16) ? ((kl & 7) >> 1) : (4 + ((kl & 7) >> 1));
  int e  = 2 * p + (kl & 1);
  int lane = (m & 15) + 16 * h;
  return ((size_t)((m >> 4) * (K >> 5) + kt) * 32 + lane) * 16 + e;
}
// B (KxN, 32x16 tiles): lane l holds column n = l&15, k order matches A (table order).
__device__ __forceinline__ size_t bpack_idx(int k, int n, int N) {
  int kt = k >> 5;
  int h  = (k >> 3) & 1;
  int e  = (k & 7) | ((k >> 1) & 8);
  int lane = (n & 15) + 16 * h;
  return ((size_t)(kt * (N >> 4) + (n >> 4)) * 32 + lane) * 16 + e;
}

// ---------------- conversion kernels (f32 row-major -> packed bf16) ----------------

__global__ void cvt_apack_kernel(const float* __restrict__ in, __bf16* __restrict__ out, int M, int K) {
  int i = blockIdx.x * blockDim.x + threadIdx.x;
  if (i >= M * K) return;
  int m = i / K, k = i - m * K;
  out[apack_idx(m, k, K)] = bf_from_f32(in[i]);
}

__global__ void cvt_bpack_kernel(const float* __restrict__ in, __bf16* __restrict__ out, int K, int N) {
  int i = blockIdx.x * blockDim.x + threadIdx.x;
  if (i >= K * N) return;
  int k = i / N, n = i - k * N;
  out[bpack_idx(k, n, N)] = bf_from_f32(in[i]);
}

__global__ void zero_u32_kernel(unsigned* __restrict__ p, int n) {
  int i = blockIdx.x * blockDim.x + threadIdx.x;
  if (i < n) p[i] = 0u;
}

__global__ void zero_f32_kernel(float* __restrict__ p, int n) {
  int i = blockIdx.x * blockDim.x + threadIdx.x;
  if (i < n) p[i] = 0.f;
}

// ---------------- WMMA GEMM: OUT = act(A(MxK) @ B(KxN) + bias [+ skip]) ----------------
// K hardcoded to 256 (8 fully-unrolled 16x16x32 WMMAs). One 16x16 tile per wave32 block.
// A/B are packed bf16 (see above); outF is f32 row-major; outBf is packed-A bf16 for chaining.
__global__ void wmma_gemm_kernel(const __bf16* __restrict__ A, const __bf16* __restrict__ Bm,
                                 const float* __restrict__ bias, const float* __restrict__ skip,
                                 float* __restrict__ outF, __bf16* __restrict__ outBf,
                                 int M, int N, int relu)
{
  const int mt   = blockIdx.x;
  const int nt   = blockIdx.y;
  const int lane = threadIdx.x & 31;
  const int half = lane >> 4;
  const int l15  = lane & 15;
  const int ntiles = N >> 4;
  const v16bf* Ap = (const v16bf*)A;
  const v16bf* Bp = (const v16bf*)Bm;
  v8f acc = {};
#pragma unroll
  for (int kt = 0; kt < KT; ++kt) {
    v16bf av = Ap[(size_t)(mt * KT + kt) * 32 + lane];      // 32B contiguous per lane
    v16bf bv = Bp[(size_t)(kt * ntiles + nt) * 32 + lane];  // 32B contiguous per lane
    acc = __builtin_amdgcn_wmma_f32_16x16x32_bf16(false, av, false, bv, (short)0, acc, false, false);
  }
  // C/D 16x16 f32 layout: VGPR r holds row (r + 8*half), col = l15
  const int n  = nt * 16 + l15;
  const float bn = bias ? bias[n] : 0.f;
#pragma unroll
  for (int r = 0; r < 8; ++r) {
    int m = mt * 16 + r + 8 * half;
    float x = acc[r] + bn;
    if (skip)  x += skip[(size_t)m * N + n];
    if (relu)  x = fmaxf(x, 0.f);
    if (outF)  outF[(size_t)m * N + n] = x;
    if (outBf) outBf[apack_idx(m, n, N)] = bf_from_f32(x);  // ready for next GEMM's A
  }
}

// ---------------- start head GEMV: out[r] = dot(X[r,:], w) + b ----------------
__global__ void gemv_bias_kernel(const float* __restrict__ X, const float* __restrict__ w,
                                 const float* __restrict__ b, float* __restrict__ out, int M, int K)
{
  int r = blockIdx.x * blockDim.x + threadIdx.x;
  if (r >= M) return;
  const float* x = X + (size_t)r * K;
  float s = b[0];
  for (int k = 0; k < K; ++k) s += x[k] * w[k];
  out[r] = s;
}

// ---------------- in-place row log-softmax (block = 256 threads, one row per block) ----------------
__global__ void row_log_softmax_kernel(float* __restrict__ x, int cols)
{
  const int tid = threadIdx.x;
  float* r = x + (size_t)blockIdx.x * cols;
  __shared__ float red[256];
  float m = -__builtin_inff();
  for (int i = tid; i < cols; i += 256) m = fmaxf(m, r[i]);
  red[tid] = m; __syncthreads();
  for (int o = 128; o > 0; o >>= 1) { if (tid < o) red[tid] = fmaxf(red[tid], red[tid + o]); __syncthreads(); }
  m = red[0]; __syncthreads();
  float s = 0.f;
  for (int i = tid; i < cols; i += 256) s += expf(r[i] - m);
  red[tid] = s; __syncthreads();
  for (int o = 128; o > 0; o >>= 1) { if (tid < o) red[tid] += red[tid + o]; __syncthreads(); }
  const float lse = m + logf(red[0]);
  __syncthreads();
  for (int i = tid; i < cols; i += 256) r[i] -= lse;
}

// ---------------- emission mask scatter + masked row LSE ----------------

__global__ void scatter_mask_kernel(const int* __restrict__ w2s, unsigned* __restrict__ bits)
{
  int idx = blockIdx.x * blockDim.x + threadIdx.x;
  if (idx >= Vv * Ss) return;
  int v = idx / Ss;
  int c = w2s[idx];
  atomicOr(&bits[(size_t)c * VW + (v >> 5)], 1u << (v & 31));
}

__global__ void masked_row_lse_kernel(const float* __restrict__ logits, const unsigned* __restrict__ bits,
                                      float* __restrict__ lse)
{
  const int c = blockIdx.x, tid = threadIdx.x;
  const float*    row = logits + (size_t)c * Vv;
  const unsigned* mb  = bits   + (size_t)c * VW;
  __shared__ float red[256];
  float m = -__builtin_inff();
  for (int v = tid; v < Vv; v += 256)
    if ((mb[v >> 5] >> (v & 31)) & 1u) m = fmaxf(m, row[v]);
  red[tid] = m; __syncthreads();
  for (int o = 128; o > 0; o >>= 1) { if (tid < o) red[tid] = fmaxf(red[tid], red[tid + o]); __syncthreads(); }
  m = red[0]; __syncthreads();
  float s = 0.f;
  for (int v = tid; v < Vv; v += 256)
    if ((mb[v >> 5] >> (v & 31)) & 1u) s += expf(row[v] - m);
  red[tid] = s; __syncthreads();
  for (int o = 128; o > 0; o >>= 1) { if (tid < o) red[tid] += red[tid + o]; __syncthreads(); }
  if (tid == 0) lse[c] = m + logf(red[0]);
}

// ---------------- clamped states + observation log-probs ----------------
// obs[b,t,s] = em_logits[c, w] - masked_lse[c], c = word2state[text[b,t], s] (always mask-true)
__global__ void build_obs_kernel(const int* __restrict__ text, const int* __restrict__ w2s,
                                 const float* __restrict__ em, const float* __restrict__ mlse,
                                 int* __restrict__ clamped, float* __restrict__ obs)
{
  int idx = blockIdx.x * blockDim.x + threadIdx.x;
  if (idx >= Bb * Tt * Ss) return;
  int s  = idx & (Ss - 1);
  int bt = idx / Ss;
  int w  = text[bt];
  int c  = w2s[w * Ss + s];
  clamped[idx] = c;
  obs[idx] = em[(size_t)c * Vv + w] - mlse[c];
}

// ---------------- HMM forward scan: one block per batch element, one thread per state ----------------
__global__ void forward_scan_kernel(const int* __restrict__ clamped, const float* __restrict__ obs,
                                    const float* __restrict__ start_ls, const float* __restrict__ trans_ls,
                                    float* __restrict__ out)
{
  const int b = blockIdx.x;
  const int j = threadIdx.x;       // state slot 0..127
  __shared__ float a[Ss];
  __shared__ int   cp[Ss];
  __shared__ float red[Ss];
  const int*   cl = clamped + (size_t)b * Tt * Ss;
  const float* ob = obs     + (size_t)b * Tt * Ss;
  int c0 = cl[j];
  cp[j] = c0;
  a[j]  = start_ls[c0] + ob[j];
  __syncthreads();
  for (int t = 0; t < Tt - 1; ++t) {
    const int cj = cl[(t + 1) * Ss + j];
    // online logsumexp over previous states i
    float m = -__builtin_inff(), s = 0.f;
    for (int i = 0; i < Ss; ++i) {
      float v = trans_ls[(size_t)cp[i] * Cc + cj] + a[i];
      if (v > m) { s = s * expf(m - v) + 1.f; m = v; }
      else       { s += expf(v - m); }
    }
    const float na = ob[(t + 1) * Ss + j] + m + logf(s);
    __syncthreads();                 // everyone done reading a[]/cp[]
    a[j]  = na;
    cp[j] = cj;
    __syncthreads();
  }
  // final logsumexp over states
  red[j] = a[j]; __syncthreads();
  for (int o = Ss / 2; o > 0; o >>= 1) { if (j < o) red[j] = fmaxf(red[j], red[j + o]); __syncthreads(); }
  const float m = red[0]; __syncthreads();
  red[j] = expf(a[j] - m); __syncthreads();
  for (int o = Ss / 2; o > 0; o >>= 1) { if (j < o) red[j] += red[j + o]; __syncthreads(); }
  if (j == 0) atomicAdd(out, m + logf(red[0]));
}

// ---------------- host orchestration ----------------

extern "C" void kernel_launch(void* const* d_in, const int* in_sizes, int n_in,
                              void* d_out, int out_size, void* d_ws, size_t ws_size,
                              hipStream_t stream)
{
  const int*   text      = (const int*)  d_in[0];
  const int*   w2s       = (const int*)  d_in[1];
  const float* start_emb = (const float*)d_in[2];
  const float* start_w1  = (const float*)d_in[3];
  const float* start_b1  = (const float*)d_in[4];
  const float* start_w2  = (const float*)d_in[5];
  const float* start_b2  = (const float*)d_in[6];
  const float* start_wo  = (const float*)d_in[7];
  const float* start_bo  = (const float*)d_in[8];
  const float* state_emb = (const float*)d_in[9];
  const float* trans_w1  = (const float*)d_in[10];
  const float* trans_b1  = (const float*)d_in[11];
  const float* trans_w2  = (const float*)d_in[12];
  const float* trans_b2  = (const float*)d_in[13];
  const float* trans_wo  = (const float*)d_in[14];
  const float* trans_bo  = (const float*)d_in[15];
  const float* pre_emb   = (const float*)d_in[16];
  const float* term_w1   = (const float*)d_in[17];
  const float* term_b1   = (const float*)d_in[18];
  const float* term_w2   = (const float*)d_in[19];
  const float* term_b2   = (const float*)d_in[20];
  const float* term_wo   = (const float*)d_in[21];
  const float* term_bo   = (const float*)d_in[22];
  (void)in_sizes; (void)n_in; (void)out_size; (void)ws_size;

  // workspace bump allocator (256B aligned)
  size_t off = 0;
  auto take = [&](size_t bytes) -> char* {
    char* p = (char*)d_ws + off;
    off = (off + bytes + 255) & ~(size_t)255;
    return p;
  };
  __bf16*   A_bf      = (__bf16*)  take((size_t)Cc * Hh * 2);   // packed-A embedding
  __bf16*   W1_bf     = (__bf16*)  take((size_t)Hh * Hh * 2);   // packed-B
  __bf16*   W2_bf     = (__bf16*)  take((size_t)Hh * Hh * 2);   // packed-B
  __bf16*   Wo_bf     = (__bf16*)  take((size_t)Hh * Vv * 2);   // packed-B (max of H*C / H*V)
  __bf16*   h1_bf     = (__bf16*)  take((size_t)Cc * Hh * 2);   // packed-A
  __bf16*   h2_bf     = (__bf16*)  take((size_t)Cc * Hh * 2);   // packed-A
  float*    h2_f      = (float*)   take((size_t)Cc * Hh * 4);
  float*    start_ls  = (float*)   take((size_t)Cc * 4);
  float*    trans_ls  = (float*)   take((size_t)Cc * Cc * 4);
  float*    em_logits = (float*)   take((size_t)Cc * Vv * 4);
  float*    mlse      = (float*)   take((size_t)Cc * 4);
  unsigned* maskbits  = (unsigned*)take((size_t)Cc * VW * 4);
  int*      clamped   = (int*)     take((size_t)Bb * Tt * Ss * 4);
  float*    obs       = (float*)   take((size_t)Bb * Tt * Ss * 4);

  auto cvtA = [&](const float* src, __bf16* dst, int M, int K) {
    cvt_apack_kernel<<<(M * K + 255) / 256, 256, 0, stream>>>(src, dst, M, K);
  };
  auto cvtB = [&](const float* src, __bf16* dst, int K, int N) {
    cvt_bpack_kernel<<<(K * N + 255) / 256, 256, 0, stream>>>(src, dst, K, N);
  };
  auto gemm = [&](const __bf16* A, const __bf16* Bm, const float* bias, const float* skip,
                  float* oF, __bf16* oB, int M, int N, int relu) {
    dim3 grid(M / 16, N / 16);
    wmma_gemm_kernel<<<grid, 32, 0, stream>>>(A, Bm, bias, skip, oF, oB, M, N, relu);
  };

  // ---- start head: logits -> log_softmax over C ----
  cvtA(start_emb, A_bf, Cc, Hh);
  cvtB(start_w1, W1_bf, Hh, Hh);
  cvtB(start_w2, W2_bf, Hh, Hh);
  gemm(A_bf, W1_bf, start_b1, nullptr, nullptr, h1_bf, Cc, Hh, 1);
  gemm(h1_bf, W2_bf, start_b2, start_emb, h2_f, nullptr, Cc, Hh, 1);
  gemv_bias_kernel<<<(Cc + 127) / 128, 128, 0, stream>>>(h2_f, start_wo, start_bo, start_ls, Cc, Hh);
  row_log_softmax_kernel<<<1, 256, 0, stream>>>(start_ls, Cc);

  // ---- transition head: (C,C) logits -> row log_softmax ----
  cvtA(state_emb, A_bf, Cc, Hh);
  cvtB(trans_w1, W1_bf, Hh, Hh);
  cvtB(trans_w2, W2_bf, Hh, Hh);
  cvtB(trans_wo, Wo_bf, Hh, Cc);
  gemm(A_bf, W1_bf, trans_b1, nullptr, nullptr, h1_bf, Cc, Hh, 1);
  gemm(h1_bf, W2_bf, trans_b2, state_emb, nullptr, h2_bf, Cc, Hh, 1);
  gemm(h2_bf, Wo_bf, trans_bo, nullptr, trans_ls, nullptr, Cc, Cc, 0);
  row_log_softmax_kernel<<<Cc, 256, 0, stream>>>(trans_ls, Cc);

  // ---- term head: (C,V) emission logits ----
  cvtA(pre_emb, A_bf, Cc, Hh);
  cvtB(term_w1, W1_bf, Hh, Hh);
  cvtB(term_w2, W2_bf, Hh, Hh);
  cvtB(term_wo, Wo_bf, Hh, Vv);
  gemm(A_bf, W1_bf, term_b1, nullptr, nullptr, h1_bf, Cc, Hh, 1);
  gemm(h1_bf, W2_bf, term_b2, pre_emb, nullptr, h2_bf, Cc, Hh, 1);
  gemm(h2_bf, Wo_bf, term_bo, nullptr, em_logits, nullptr, Cc, Vv, 0);

  // ---- emission mask + masked row logsumexp ----
  zero_u32_kernel<<<(Cc * VW + 255) / 256, 256, 0, stream>>>(maskbits, Cc * VW);
  scatter_mask_kernel<<<(Vv * Ss + 255) / 256, 256, 0, stream>>>(w2s, maskbits);
  masked_row_lse_kernel<<<Cc, 256, 0, stream>>>(em_logits, maskbits, mlse);

  // ---- clamped states + observation scores ----
  build_obs_kernel<<<(Bb * Tt * Ss + 255) / 256, 256, 0, stream>>>(text, w2s, em_logits, mlse, clamped, obs);

  // ---- forward scan and batch sum ----
  zero_f32_kernel<<<1, 1, 0, stream>>>((float*)d_out, 1);
  forward_scan_kernel<<<Bb, Ss, 0, stream>>>(clamped, obs, start_ls, trans_ls, (float*)d_out);
}
